// DenseDeformableModule_7121055776937
// MI455X (gfx1250) — matrix operand order repaired and checked by reading, not compiled, not run on hardware
//
#include <hip/hip_runtime.h>
#include <hip/hip_bf16.h>

typedef __attribute__((ext_vector_type(16))) __bf16 v16bf;
typedef __attribute__((ext_vector_type(8)))  float  v8f;
typedef __attribute__((ext_vector_type(4)))  unsigned int v4u;

#define CIN   128
#define OOUT  128
#define HH    128
#define WW    128
#define NB    4
#define HWSZ  (HH*WW)        // 16384
#define KRED  (9*CIN)        // 1152
#define NCHUNK (KRED/32)     // 36

// LDS B-chunk staging: 256 entries (nt*32+lane) of 32B data, padded to 48B
// stride (16B aligned; 12-dword lane stride -> only 2-way ds bank conflicts).
#define BENT_US 24           // ushorts per entry (48 B)

__device__ __forceinline__ __bf16 f2bf(float f) {
    unsigned u = __builtin_bit_cast(unsigned, f);
    u += 0x7FFFu + ((u >> 16) & 1u);            // round-to-nearest-even
    unsigned short h = (unsigned short)(u >> 16);
    return __builtin_bit_cast(__bf16, h);
}

// Pack two f32 -> packed bf16x2 (RNE), one VALU op when HW cvt available.
__device__ __forceinline__ unsigned cvt_pk_bf16(float s0, float s1) {
#if __has_builtin(__builtin_amdgcn_cvt_pk_bf16_f32)
    typedef __attribute__((ext_vector_type(2))) __bf16 v2bf;
    v2bf pk = __builtin_amdgcn_cvt_pk_bf16_f32(s0, s1);
    return __builtin_bit_cast(unsigned, pk);
#else
    unsigned r;
    asm("v_cvt_pk_bf16_f32 %0, %1, %2" : "=v"(r) : "v"(s0), "v"(s1));
    return r;
#endif
}

__device__ __forceinline__ void wait_asynccnt0() {
#if __has_builtin(__builtin_amdgcn_s_wait_asynccnt)
    __builtin_amdgcn_s_wait_asynccnt(0);
#else
    asm volatile("s_wait_asynccnt 0x0" ::: "memory");
#endif
}

// ---------------------------------------------------------------------------
// Kernel 1: repack w_conv (O,C,3,3) f32 -> bf16 B-fragments.
// k_red = tap*128 + c, chunked by 32 (36 chunks), N tiled by 16 (8 tiles).
// B-fragment layout used here (32x16 K x N, 16-bit, contiguous k):
//   lane L<16 : N=L,    elements 0..15 hold K = 0..15
//   lane L>=16: N=L-16, elements 0..15 hold K = 16..31
// Dense global layout: packed[((kc*8 + nt)*32 + lane)*16 + i]
// ---------------------------------------------------------------------------
__global__ void pack_wconv_kernel(const float* __restrict__ w_conv,
                                  unsigned short* __restrict__ packed) {
    int idx = blockIdx.x * blockDim.x + threadIdx.x;   // 147456
    if (idx >= NCHUNK * 8 * 32 * 16) return;
    int i  = idx & 15;
    int L  = (idx >> 4) & 31;
    int nt = (idx >> 9) & 7;
    int kc = idx >> 12;

    int tap = kc >> 2;
    int c0  = (kc & 3) * 32;
    int kstart = (L < 16) ? 0 : 16;
    int ch = c0 + kstart + i;
    int o  = nt * 16 + (L & 15);
    int ky = tap / 3, kx = tap % 3;
    float v = w_conv[((o * CIN + ch) * 3 + ky) * 3 + kx];
    packed[idx] = __builtin_bit_cast(unsigned short, f2bf(v));
}

// ---------------------------------------------------------------------------
// Kernel 2: offset conv (fp32 exact; offsets feed sampling coordinates).
// Thread per pixel, 18 accumulators; weight indices are wave-uniform ->
// scalar loads; x reads coalesced in W and L1/L2 resident.
// ---------------------------------------------------------------------------
__global__ void offset_conv_kernel(const float* __restrict__ x,
                                   const float* __restrict__ w_off,
                                   float* __restrict__ offbuf) {
    int p = blockIdx.x * blockDim.x + threadIdx.x;
    int b  = p >> 14;
    int hw = p & (HWSZ - 1);
    int h  = hw >> 7;
    int w  = hw & (WW - 1);

    const float* xb = x + (size_t)b * CIN * HWSZ;
    float acc[18];
#pragma unroll
    for (int oc = 0; oc < 18; ++oc) acc[oc] = 0.f;

#pragma unroll 1
    for (int c = 0; c < CIN; ++c) {
        const float* xc = xb + (size_t)c * HWSZ;
        float xv[9];
#pragma unroll
        for (int t = 0; t < 9; ++t) {
            int yy = h - 1 + t / 3;
            int xx = w - 1 + t % 3;
            bool ok = (yy >= 0) & (yy < HH) & (xx >= 0) & (xx < WW);
            xv[t] = ok ? xc[yy * WW + xx] : 0.f;
        }
#pragma unroll
        for (int oc = 0; oc < 18; ++oc) {
            const float* wp = w_off + ((size_t)oc * CIN + c) * 9;
            float s = acc[oc];
#pragma unroll
            for (int t = 0; t < 9; ++t) s += xv[t] * wp[t];
            acc[oc] = s;
        }
    }
#pragma unroll
    for (int oc = 0; oc < 18; ++oc)
        offbuf[((size_t)b * 18 + oc) * HWSZ + hw] = acc[oc];
}

// ---------------------------------------------------------------------------
// Kernel 3: fused deformable gather + bf16 WMMA GEMM.
// Block = 128 threads = 4 waves; wave = 16 consecutive W-pixels x N=128.
// Per 32-wide K chunk: the block DMAs the 8 KB B-chunk once into LDS via
// GLOBAL_LOAD_ASYNC_TO_LDS_B128 (overlapped with the bilinear gather that
// builds the A fragment), then each wave runs 8 WMMAs with software-pipelined
// fragment loads from LDS. Output transposed through padded LDS for
// coalesced streaming NCHW stores.
// ---------------------------------------------------------------------------
__global__ void deform_main_kernel(const float* __restrict__ x,
                                   const float* __restrict__ offbuf,
                                   const unsigned short* __restrict__ packed,
                                   float* __restrict__ out) {
    __shared__ unsigned short bbuf[256 * BENT_US];   // 12,288 B
    __shared__ float lds_out[4][16][129];            // 33,024 B

    const int tid  = threadIdx.x;
    const int wv   = tid >> 5;
    const int lane = tid & 31;
    const int lm   = lane & 15;
    const int hi   = lane >> 4;
    const int kb   = hi ? 8 : 0;         // unused; A-fragment K base is hi*16

    const int p  = blockIdx.x * 64 + wv * 16 + lm;
    const int b  = p >> 14;
    const int hw = p & (HWSZ - 1);
    const int h  = hw >> 7;
    const int w  = hw & (WW - 1);

    const float* xb = x + (size_t)b * CIN * HWSZ;

    // Per-thread async-copy slots: entries e = tid and tid+128, 32B each.
    const unsigned lds0 = (unsigned)(size_t)&bbuf[(unsigned)tid * BENT_US];
    const unsigned lds1 = (unsigned)(size_t)&bbuf[(unsigned)(tid + 128) * BENT_US];

    union Frag { struct { v4u lo, hi; } u; v16bf v; };
    union AFrag { unsigned u[8]; v16bf v; };

    v8f acc[8] = {};

#pragma unroll 1
    for (int tap = 0; tap < 9; ++tap) {
        const int ky = tap / 3, kx = tap % 3;
        const float dy = offbuf[((size_t)b * 18 + tap * 2 + 0) * HWSZ + hw];
        const float dx = offbuf[((size_t)b * 18 + tap * 2 + 1) * HWSZ + hw];
        const float py = (float)(h - 1 + ky) + dy;
        const float px = (float)(w - 1 + kx) + dx;
        const float fy = floorf(py), fx = floorf(px);
        const float wy1 = py - fy, wy0 = 1.f - wy1;
        const float wx1 = px - fx, wx0 = 1.f - wx1;
        const int iy0 = (int)fy, ix0 = (int)fx;
        const int iy1 = iy0 + 1, ix1 = ix0 + 1;
        const float my0 = (iy0 >= 0 && iy0 < HH) ? 1.f : 0.f;
        const float my1 = (iy1 >= 0 && iy1 < HH) ? 1.f : 0.f;
        const float mx0 = (ix0 >= 0 && ix0 < WW) ? 1.f : 0.f;
        const float mx1 = (ix1 >= 0 && ix1 < WW) ? 1.f : 0.f;
        const float w00 = wy0 * wx0 * my0 * mx0;
        const float w01 = wy0 * wx1 * my0 * mx1;
        const float w10 = wy1 * wx0 * my1 * mx0;
        const float w11 = wy1 * wx1 * my1 * mx1;
        const int cy0 = min(max(iy0, 0), HH - 1);
        const int cy1 = min(max(iy1, 0), HH - 1);
        const int cx0 = min(max(ix0, 0), WW - 1);
        const int cx1 = min(max(ix1, 0), WW - 1);
        const int o00 = cy0 * WW + cx0, o01 = cy0 * WW + cx1;
        const int o10 = cy1 * WW + cx0, o11 = cy1 * WW + cx1;

#pragma unroll 1
        for (int sub = 0; sub < 4; ++sub) {
            const int c0 = sub * 32;
            const int kc = tap * 4 + sub;

            // All waves done reading the previous B chunk before overwrite.
            __syncthreads();

            // Kick off the async DMA of this chunk (4x b128 per thread).
            {
                const unsigned short* g0 = packed + ((size_t)kc * 256 + tid) * 16;
                const unsigned short* g1 = g0 + 128 * 16;
                asm volatile("global_load_async_to_lds_b128 %0, %1, off"
                             :: "v"(lds0), "v"(g0) : "memory");
                asm volatile("global_load_async_to_lds_b128 %0, %1, off"
                             :: "v"(lds0 + 16u), "v"(g0 + 8) : "memory");
                asm volatile("global_load_async_to_lds_b128 %0, %1, off"
                             :: "v"(lds1), "v"(g1) : "memory");
                asm volatile("global_load_async_to_lds_b128 %0, %1, off"
                             :: "v"(lds1 + 16u), "v"(g1 + 8) : "memory");
            }

            // Bilinear gather -> A fragment (overlaps the async DMA).
            // A layout (matches packed B): elem i -> K = hi*16 + i
            AFrag a;
#pragma unroll
            for (int i = 0; i < 16; i += 2) {
                const int kl0 = hi * 16 + i;
                const int kl1 = hi * 16 + i + 1;
                const float* pc0 = xb + (size_t)(c0 + kl0) * HWSZ;
                const float* pc1 = xb + (size_t)(c0 + kl1) * HWSZ;
                float s0 = w00 * pc0[o00] + w01 * pc0[o01]
                         + w10 * pc0[o10] + w11 * pc0[o11];
                float s1 = w00 * pc1[o00] + w01 * pc1[o01]
                         + w10 * pc1[o10] + w11 * pc1[o11];
                a.u[i >> 1] = cvt_pk_bf16(s0, s1);
            }

            wait_asynccnt0();
            __syncthreads();

            // 8 WMMAs with software-pipelined LDS fragment loads: the loads
            // for fragment nt+1 issue before WMMA nt, so s_wait_dscnt can
            // tolerate outstanding loads and LDS latency hides under WMMA.
            Frag cur;
            cur.u.lo = *(const v4u*)&bbuf[(unsigned)lane * BENT_US];
            cur.u.hi = *(const v4u*)&bbuf[(unsigned)lane * BENT_US + 8];
#pragma unroll
            for (int nt = 0; nt < 8; ++nt) {
                Frag nxt;
                if (nt < 7) {
                    const unsigned e = (unsigned)((nt + 1) * 32 + lane);
                    nxt.u.lo = *(const v4u*)&bbuf[e * BENT_US];
                    nxt.u.hi = *(const v4u*)&bbuf[e * BENT_US + 8];
                }
                acc[nt] = __builtin_amdgcn_wmma_f32_16x16x32_bf16(
                    false, a.v, false, cur.v, (short)0, acc[nt], false, false);
                if (nt < 7) cur = nxt;
            }
        }
    }

    // D layout: acc[nt] VGPR r -> M = r + hi*8, N = nt*16 + lm.
    __syncthreads();
#pragma unroll
    for (int nt = 0; nt < 8; ++nt) {
#pragma unroll
        for (int r = 0; r < 8; ++r) {
            lds_out[wv][r + hi * 8][nt * 16 + lm] = acc[nt][r];
        }
    }
    __syncthreads();

    // Coalesced streaming NCHW stores: lanes sweep 16 consecutive W-pixels.
#pragma unroll 4
    for (int og = 0; og < 64; ++og) {
        const int o = og * 2 + hi;
        const float v = lds_out[wv][lm][o];
        __builtin_nontemporal_store(v, &out[(((size_t)b * OOUT + o) << 14) + hw]);
    }
}

extern "C" void kernel_launch(void* const* d_in, const int* in_sizes, int n_in,
                              void* d_out, int out_size, void* d_ws, size_t ws_size,
                              hipStream_t stream) {
    const float* x      = (const float*)d_in[0];   // (4,128,128,128)
    const float* w_off  = (const float*)d_in[1];   // (18,128,3,3)
    const float* w_conv = (const float*)d_in[2];   // (128,128,3,3)
    float* out = (float*)d_out;                    // (4,128,128,128)

    float* offbuf = (float*)d_ws;                                  // 4.72 MB
    unsigned short* packed =
        (unsigned short*)((char*)d_ws + (size_t)NB * 18 * HWSZ * sizeof(float));

    pack_wconv_kernel<<<(NCHUNK * 8 * 32 * 16 + 255) / 256, 256, 0, stream>>>(
        w_conv, packed);
    offset_conv_kernel<<<(NB * HWSZ) / 256, 256, 0, stream>>>(x, w_off, offbuf);
    deform_main_kernel<<<(NB * HWSZ) / 64, 128, 0, stream>>>(x, offbuf, packed, out);
}
